// SchNet_73555609911514
// MI455X (gfx1250) — matrix operand order repaired and checked by reading, not compiled, not run on hardware
//
#include <hip/hip_runtime.h>
#include <hip/hip_bf16.h>
#include <math.h>

typedef _Float16 v16h __attribute__((ext_vector_type(16)));
typedef _Float16 v8h  __attribute__((ext_vector_type(8)));
typedef float    v8f  __attribute__((ext_vector_type(8)));

#define BATCH 8
#define NA    256
#define NN    255
#define NBF   128
#define NGAUSS 25
#define NITER 3

#define GWIDTH  (5.0f/24.0f)
#define GCOEFF  (-0.5f/(GWIDTH*GWIDTH))
#define PI_OVER_CUT 0.62831853071f

// f16 transposed-weight workspace layout (in half elements, per interaction i)
#define WT_FW1T   0        // [128][32]  (gaussian dim padded 25->32)
#define WT_FW2T   4096     // [128][128] fw2t[g][f] = fw2[f][g]
#define WT_IN2FT  20480    // [128][128] in2ft[f][d] = in2f[d][f]
#define WT_F2OUTT 36864    // [128][128] f2outt[d][f] = f2out_w[f][d]
#define WT_DENSET 53248    // [128][128] denset[e][d] = dense_w[d][e]
#define WT_STRIDE 69632
#define WS_WT_BYTES ((size_t)WT_STRIDE * NITER * 2)
#define WS_Y_OFF    WS_WT_BYTES
#define WS_Y_BYTES  ((size_t)BATCH * NA * NBF * 4)
#define WS_AGG_OFF  (WS_Y_OFF + WS_Y_BYTES)

// ---------- helpers ----------
__device__ __forceinline__ v8f wmma16(v16h a, v16h b, v8f c) {
  return __builtin_amdgcn_wmma_f32_16x16x32_f16(false, a, false, b,
                                                (short)0, c, false, false);
}

__device__ __forceinline__ v8f zero8() {
  v8f z;
#pragma unroll
  for (int i = 0; i < 8; ++i) z[i] = 0.f;
  return z;
}

__device__ __forceinline__ v16h pack16(v8h lo, v8h hi) {
  v16h r;
#pragma unroll
  for (int i = 0; i < 8; ++i) { r[i] = lo[i]; r[8 + i] = hi[i]; }
  return r;
}

// A-operand (16-bit 16x32): lane holds row M=L&15, K = (L>>4)*8 + {0..7, 16..23}
__device__ __forceinline__ v16h loadA_f32(const float* row, int k0) {
  v16h r;
#pragma unroll
  for (int i = 0; i < 8; ++i) {
    r[i]     = (_Float16)row[k0 + i];
    r[8 + i] = (_Float16)row[k0 + 16 + i];
  }
  return r;
}

__device__ __forceinline__ v16h loadA_h(const _Float16* row, int k0) {
  v8h lo = *(const v8h*)(row + k0);
  v8h hi = *(const v8h*)(row + k0 + 16);
  return pack16(lo, hi);
}

// B-operand (16-bit 32x16, stored as Bt[n][k] row-major):
// lane holds col N=L&15, K = (L>>4)*8 + {0..7, 16..23}
__device__ __forceinline__ v16h loadB_h(const _Float16* row, int k0) {
  v8h lo = *(const v8h*)(row + k0);
  v8h hi = *(const v8h*)(row + k0 + 16);
  return pack16(lo, hi);
}

// shifted softplus, numerically stable
__device__ __forceinline__ float sspf(float x) {
  float e = __expf(-fabsf(x));
  return fmaxf(x, 0.f) + __logf(1.f + e) - 0.69314718f;
}

// ---------- kernel 0: weight convert + transpose to f16 ----------
__global__ void k_prep(const float* __restrict__ fw1, const float* __restrict__ fw2,
                       const float* __restrict__ in2f, const float* __restrict__ f2out,
                       const float* __restrict__ dense, _Float16* __restrict__ wt) {
  int idx = blockIdx.x * 256 + threadIdx.x;
  if (idx >= WT_STRIDE * NITER) return;
  int i = idx / WT_STRIDE;
  int r = idx - i * WT_STRIDE;
  float v;
  if (r < WT_FW2T) {                       // fw1t[f][g] = fw1[g][f], pad g>=25
    int f = r >> 5, g = r & 31;
    v = (g < NGAUSS) ? fw1[(i * NGAUSS + g) * NBF + f] : 0.f;
  } else if (r < WT_IN2FT) {               // fw2t[g][f] = fw2[f][g]
    int rr = r - WT_FW2T; int g = rr >> 7, f = rr & 127;
    v = fw2[((size_t)i * NBF + f) * NBF + g];
  } else if (r < WT_F2OUTT) {              // in2ft[f][d] = in2f[d][f]
    int rr = r - WT_IN2FT; int f = rr >> 7, d = rr & 127;
    v = in2f[((size_t)i * NBF + d) * NBF + f];
  } else if (r < WT_DENSET) {              // f2outt[d][f] = f2out_w[f][d]
    int rr = r - WT_F2OUTT; int d = rr >> 7, f = rr & 127;
    v = f2out[((size_t)i * NBF + f) * NBF + d];
  } else {                                 // denset[e][d] = dense_w[d][e]
    int rr = r - WT_DENSET; int e = rr >> 7, d = rr & 127;
    v = dense[((size_t)i * NBF + d) * NBF + e];
  }
  wt[idx] = (_Float16)v;
}

// ---------- kernel 1: embedding gather into x ----------
__global__ void k_embed(const int* __restrict__ zn, const float* __restrict__ emb,
                        float* __restrict__ x) {
  int idx = blockIdx.x * 256 + threadIdx.x;    // BATCH*NA*NBF
  int ba = idx >> 7, f = idx & 127;
  x[idx] = emb[zn[ba] * NBF + f];
}

// ---------- kernel 2: y = x @ in2f  (one 16x16 tile per wave) ----------
__global__ void __launch_bounds__(256)
k_in2f(const float* __restrict__ x, const _Float16* __restrict__ in2ft,
       float* __restrict__ y) {
  int lane = threadIdx.x & 31, wid = threadIdx.x >> 5;
  int gw = blockIdx.x * 8 + wid;
  int mt = gw >> 3, nt = gw & 7;
  int col = lane & 15, hf = lane >> 4;
  int row = mt * 16 + col;
  int ncol = nt * 16 + col;
  const float*    xr = x     + (size_t)row  * NBF;
  const _Float16* br = in2ft + (size_t)ncol * NBF;
  v8f acc = zero8();
#pragma unroll
  for (int kc = 0; kc < 4; ++kc) {
    v16h a  = loadA_f32(xr, kc * 32 + hf * 8);
    v16h bm = loadB_h(br, kc * 32 + hf * 8);
    acc = wmma16(a, bm, acc);
  }
#pragma unroll
  for (int r = 0; r < 8; ++r)
    y[(size_t)(mt * 16 + r + hf * 8) * NBF + ncol] = acc[r];
}

// ---------- kernel 3: fused cfconv (filter net + gather + aggregate) ----------
// one workgroup per (b,a); 8 waves x 2 neighbor-tiles of 16
__global__ void __launch_bounds__(256)
k_cfconv(const float* __restrict__ pos, const int* __restrict__ nbrs,
         const float* __restrict__ mask, const float* __restrict__ y,
         const _Float16* __restrict__ fw1t, const _Float16* __restrict__ fw2t,
         const float* __restrict__ fb1, const float* __restrict__ fb2,
         float* __restrict__ agg) {
  __shared__ __align__(16) _Float16 sW1[8][16 * NBF];   // 32 KB
  __shared__ float sAgg[8][NBF];                        // 4 KB
  int ba = blockIdx.x;
  int b = ba >> 8, a = ba & 255;
  int lane = threadIdx.x & 31, wid = threadIdx.x >> 5;
  int m = lane & 15, hf = lane >> 4;

  // warm the near cache with the weight streams (global_prefetch_b8)
  __builtin_prefetch((const void*)(fw2t + (size_t)threadIdx.x * 64), 0, 1);
  __builtin_prefetch((const void*)(fw1t + (size_t)threadIdx.x * 16), 0, 1);

  const float* posB = pos + (size_t)b * NA * 3;
  float px = posB[a * 3], py = posB[a * 3 + 1], pz = posB[a * 3 + 2];
  float aggreg[8];
#pragma unroll
  for (int q = 0; q < 8; ++q) aggreg[q] = 0.f;

  for (int tt = 0; tt < 2; ++tt) {
    int t = wid + tt * 8;              // neighbor tile 0..15
    int nidx = t * 16 + m;
    int j = 0; float rij = 0.f, cc = 0.f;
    if (nidx < NN) {
      size_t e = (size_t)ba * NN + nidx;
      j = nbrs[e];
      float mk = mask[e];
      float dx = posB[j * 3] - px, dy = posB[j * 3 + 1] - py, dz = posB[j * 3 + 2] - pz;
      rij = sqrtf(dx * dx + dy * dy + dz * dz + 1e-12f) * mk;
      float cu = (rij < 5.0f) ? 0.5f * (__cosf(rij * PI_OVER_CUT) + 1.0f) : 0.f;
      cc = cu * mk;                    // cutoff * neighbor_mask folded together
    }
    // Gaussian features directly in WMMA A layout (K padded 25->32)
    int kb = hf * 8;
    v16h a1;
#pragma unroll
    for (int u = 0; u < 8; ++u) {
      int g0 = kb + u;
      float d0 = rij - (float)g0 * GWIDTH;
      a1[u] = (_Float16)__expf(GCOEFF * d0 * d0);
      int g1 = kb + 16 + u;
      float d1 = rij - (float)g1 * GWIDTH;
      a1[8 + u] = (_Float16)((g1 < NGAUSS) ? __expf(GCOEFF * d1 * d1) : 0.f);
    }
    // GEMM1: [16,32] x [32,128] + bias + ssp -> sW1 (f16)
    _Float16* w1 = &sW1[wid][0];
#pragma unroll
    for (int ntile = 0; ntile < 8; ++ntile) {
      int f = ntile * 16 + m;
      v16h bm = loadB_h(fw1t + f * 32, hf * 8);
      v8f c = wmma16(a1, bm, zero8());
      float bias = fb1[f];
#pragma unroll
      for (int r = 0; r < 8; ++r)
        w1[(r + hf * 8) * NBF + f] = (_Float16)sspf(c[r] + bias);
    }
    // GEMM2: [16,128] x [128,128]  (LDS in-order per wave; no barrier needed)
    v8f acc2[8];
#pragma unroll
    for (int q = 0; q < 8; ++q) acc2[q] = zero8();
#pragma unroll
    for (int kc = 0; kc < 4; ++kc) {
      v16h a2 = loadA_h(w1 + m * NBF, kc * 32 + hf * 8);
#pragma unroll
      for (int ntile = 0; ntile < 8; ++ntile) {
        int g = ntile * 16 + m;
        v16h bm = loadB_h(fw2t + g * NBF, kc * 32 + hf * 8);
        acc2[ntile] = wmma16(a2, bm, acc2[ntile]);
      }
    }
    // epilogue: hoist per-row shuffles out of the feature-tile loop
    float ccm[8];
    const float* yrow[8];
    const float* yB = y + (size_t)b * NA * NBF;
#pragma unroll
    for (int r = 0; r < 8; ++r) {
      int mrow = r + hf * 8;
      ccm[r] = __shfl(cc, mrow, 32);
      int jj  = __shfl(j, mrow, 32);
      yrow[r] = yB + (size_t)jj * NBF;
    }
#pragma unroll
    for (int ntile = 0; ntile < 8; ++ntile) {
      int g = ntile * 16 + m;
      float bias = fb2[g];
      float p = 0.f;
#pragma unroll
      for (int r = 0; r < 8; ++r) {
        float yv = yrow[r][g];
        p += (acc2[ntile][r] + bias) * ccm[r] * yv;
      }
      p += __shfl_xor(p, 16, 32);      // combine row halves
      aggreg[ntile] += p;
    }
  }
  if (lane < 16) {
#pragma unroll
    for (int q = 0; q < 8; ++q) sAgg[wid][q * 16 + lane] = aggreg[q];
  }
  __syncthreads();
  if (threadIdx.x < NBF) {
    float s = 0.f;
#pragma unroll
    for (int w = 0; w < 8; ++w) s += sAgg[w][threadIdx.x];
    agg[(size_t)ba * NBF + threadIdx.x] = s;
  }
}

// ---------- kernel 4: v = ssp(agg@f2out+b)@dense+b ; x += v ----------
__global__ void __launch_bounds__(256)
k_out(const float* __restrict__ agg, const _Float16* __restrict__ f2outt,
      const _Float16* __restrict__ denset, const float* __restrict__ f2out_b,
      const float* __restrict__ dense_b, float* __restrict__ x) {
  __shared__ __align__(16) _Float16 sT[8][16 * NBF];    // 32 KB
  int lane = threadIdx.x & 31, wid = threadIdx.x >> 5;
  int mt = blockIdx.x * 8 + wid;
  int m = lane & 15, hf = lane >> 4;
  const float* ar = agg + (size_t)(mt * 16 + m) * NBF;
  _Float16* t1 = &sT[wid][0];
  // preload A chunks once
  v16h aA[4];
#pragma unroll
  for (int kc = 0; kc < 4; ++kc) aA[kc] = loadA_f32(ar, kc * 32 + hf * 8);
  // GEMM A
#pragma unroll
  for (int ntile = 0; ntile < 8; ++ntile) {
    int d = ntile * 16 + m;
    v8f acc = zero8();
#pragma unroll
    for (int kc = 0; kc < 4; ++kc) {
      v16h bv = loadB_h(f2outt + d * NBF, kc * 32 + hf * 8);
      acc = wmma16(aA[kc], bv, acc);
    }
    float bias = f2out_b[d];
#pragma unroll
    for (int r = 0; r < 8; ++r)
      t1[(r + hf * 8) * NBF + d] = (_Float16)sspf(acc[r] + bias);
  }
  // GEMM B + residual update
  v8f acc2[8];
#pragma unroll
  for (int q = 0; q < 8; ++q) acc2[q] = zero8();
#pragma unroll
  for (int kc = 0; kc < 4; ++kc) {
    v16h av = loadA_h(t1 + m * NBF, kc * 32 + hf * 8);
#pragma unroll
    for (int ntile = 0; ntile < 8; ++ntile) {
      v16h bv = loadB_h(denset + (ntile * 16 + m) * NBF, kc * 32 + hf * 8);
      acc2[ntile] = wmma16(av, bv, acc2[ntile]);
    }
  }
#pragma unroll
  for (int ntile = 0; ntile < 8; ++ntile) {
    int e = ntile * 16 + m;
    float bias = dense_b[e];
#pragma unroll
    for (int r = 0; r < 8; ++r) {
      size_t idx = (size_t)(mt * 16 + r + hf * 8) * NBF + e;
      x[idx] += acc2[ntile][r] + bias;
    }
  }
}

extern "C" void kernel_launch(void* const* d_in, const int* in_sizes, int n_in,
                              void* d_out, int out_size, void* d_ws, size_t ws_size,
                              hipStream_t stream) {
  (void)in_sizes; (void)n_in; (void)out_size; (void)ws_size;
  const int*   zn     = (const int*)  d_in[0];
  const float* pos    = (const float*)d_in[1];
  const int*   nbrs   = (const int*)  d_in[2];
  const float* mask   = (const float*)d_in[3];
  const float* emb    = (const float*)d_in[4];
  const float* fw1    = (const float*)d_in[5];
  const float* fb1    = (const float*)d_in[6];
  const float* fw2    = (const float*)d_in[7];
  const float* fb2    = (const float*)d_in[8];
  const float* in2f   = (const float*)d_in[9];
  const float* f2outw = (const float*)d_in[10];
  const float* f2outb = (const float*)d_in[11];
  const float* densew = (const float*)d_in[12];
  const float* denseb = (const float*)d_in[13];
  float* x = (float*)d_out;                    // x lives in d_out throughout
  char* ws = (char*)d_ws;
  _Float16* wt  = (_Float16*)ws;
  float*    y   = (float*)(ws + WS_Y_OFF);
  float*    agg = (float*)(ws + WS_AGG_OFF);

  k_prep<<<(WT_STRIDE * NITER + 255) / 256, 256, 0, stream>>>(fw1, fw2, in2f, f2outw, densew, wt);
  k_embed<<<(BATCH * NA * NBF) / 256, 256, 0, stream>>>(zn, emb, x);
  for (int i = 0; i < NITER; ++i) {
    const _Float16* wti = wt + (size_t)i * WT_STRIDE;
    k_in2f<<<128, 256, 0, stream>>>(x, wti + WT_IN2FT, y);
    k_cfconv<<<BATCH * NA, 256, 0, stream>>>(pos, nbrs, mask, y,
        wti + WT_FW1T, wti + WT_FW2T, fb1 + i * NBF, fb2 + i * NBF, agg);
    k_out<<<BATCH * NA / 128, 256, 0, stream>>>(agg, wti + WT_F2OUTT, wti + WT_DENSET,
        f2outb + i * NBF, denseb + i * NBF, x);
  }
}
